// SimplifiedHAFE_Baseline_59554016526894
// MI455X (gfx1250) — compile-verified, matrix operand order checked by reading, not run on hardware
//
#include <hip/hip_runtime.h>
#include <stdint.h>

#define NN 50000
#define NE 800000
#define DIN 768
#define HDIM 64
#define NH1 4
#define HC1 256

typedef __attribute__((ext_vector_type(16))) __bf16 v16bf;
typedef __attribute__((ext_vector_type(8)))  float  v8f;

// ---------- helpers ----------
__device__ __forceinline__ unsigned f2ord(float f) {
    unsigned u = __builtin_bit_cast(unsigned, f);
    return (u & 0x80000000u) ? ~u : (u | 0x80000000u);
}
__device__ __forceinline__ float ord2f(unsigned u) {
    u = (u & 0x80000000u) ? (u & 0x7FFFFFFFu) : ~u;
    return __builtin_bit_cast(float, u);
}
__device__ __forceinline__ float lrelu(float x) { return x > 0.f ? x : 0.2f * x; }

// CDNA5 WMMA 16x16x32 bf16 fragment K-offset for element i of v16bf, half lh
__device__ __forceinline__ int wmma_koff(int i, int lh) {
    int v = i >> 1, j = i & 1;
    int base = (v < 4) ? (2 * v) : (16 + 2 * (v - 4));
    return base + j + 8 * lh;
}

// ---------- weight pre-pack: fp32 [KDIM,NDIM] -> bf16 fragment order ----------
// Wp[((ct*NKB + kb)*32 + lane)*16 + i] = bf16(W[(kb*32 + koff(i,lane>>4))*NDIM + ct*16 + (lane&15)])
__global__ void pack_weights_kernel(const float* __restrict__ W, __bf16* __restrict__ Wp,
                                    int KDIM, int NDIM)
{
    int idx = blockIdx.x * blockDim.x + threadIdx.x;
    if (idx >= KDIM * NDIM) return;
    int i = idx & 15;
    int rest = idx >> 4;
    int lane = rest & 31;
    rest >>= 5;
    int nkb = KDIM >> 5;
    int kb = rest % nkb;
    int ct = rest / nkb;
    int k = kb * 32 + wmma_koff(i, lane >> 4);
    int col = ct * 16 + (lane & 15);
    Wp[idx] = (__bf16)W[(size_t)k * NDIM + col];
}

// ---------- stage 16 x KDIM strip of A into LDS in WMMA fragment order ----------
template <int KDIM>
__device__ __forceinline__ void stage_A(const float* __restrict__ A, __bf16* As,
                                        int rowBase, int M, int tid)
{
    for (int idx = tid; idx < 16 * KDIM; idx += 128) {
        int i = idx & 15;
        int rest = idx >> 4;
        int ln = rest & 31;
        int kb = rest >> 5;
        int row = ln & 15;
        int k = kb * 32 + wmma_koff(i, ln >> 4);
        int grow = rowBase + row; if (grow >= M) grow = M - 1;
        As[idx] = (__bf16)A[(size_t)grow * KDIM + k];
    }
}

// ---------- WMMA GEMM: C[M,NDIM] = epi(A[M,KDIM] @ W[KDIM,NDIM] + bias) ----------
// Wp is bf16 pre-packed in fragment order. A staged in LDS in fragment order.
// EPI: 0 = bias (optional), 1 = bias + relu, 2 = bias + type_emb + pos outer prod
template <int KDIM, int NDIM, int EPI>
__global__ __launch_bounds__(128) void wmma_gemm_kernel(
    const float* __restrict__ A, const __bf16* __restrict__ Wp,
    const float* __restrict__ bias, float* __restrict__ C, int M,
    const int* __restrict__ ntypes, const float* __restrict__ temb,
    const float* __restrict__ pos, const float* __restrict__ posw,
    const float* __restrict__ posb)
{
    constexpr int NKB = KDIM / 32;
    __shared__ __align__(32) __bf16 As[NKB * 32 * 16];   // fragment-ordered A strip
    const int tid = threadIdx.x;
    const int rowBase = blockIdx.x * 16;

    stage_A<KDIM>(A, As, rowBase, M, tid);
    __syncthreads();

    const int wid  = tid >> 5;
    const int lane = tid & 31;
    const int lr   = lane & 15;
    const int lh   = lane >> 4;
    const int ct   = blockIdx.y * 4 + wid;
    const int col  = ct * 16 + lr;

    const __bf16* wp = Wp + ((size_t)ct * NKB * 32 + lane) * 16;

    v8f acc = {};
#pragma unroll
    for (int kb = 0; kb < NKB; ++kb) {
        v16bf a = *(const v16bf*)&As[(kb * 32 + lane) * 16];
        v16bf b = *(const v16bf*)&wp[(size_t)kb * 32 * 16];
        acc = __builtin_amdgcn_wmma_f32_16x16x32_bf16(
            false, a, false, b, (short)0, acc, false, false);
    }

    const float bcol = bias ? bias[col] : 0.0f;
    if (rowBase + 16 <= M) {        // fast path: whole tile in range (wave-uniform)
#pragma unroll
        for (int r = 0; r < 8; ++r) {
            int row = rowBase + r + 8 * lh;
            float v = acc[r] + bcol;
            if (EPI == 1) v = v > 0.f ? v : 0.f;
            if (EPI == 2)
                v += temb[ntypes[row] * NDIM + col] + pos[row] * posw[col] + posb[col];
            C[(size_t)row * NDIM + col] = v;
        }
    } else {
#pragma unroll
        for (int r = 0; r < 8; ++r) {
            int row = rowBase + r + 8 * lh;
            if (row < M) {
                float v = acc[r] + bcol;
                if (EPI == 1) v = v > 0.f ? v : 0.f;
                if (EPI == 2)
                    v += temb[ntypes[row] * NDIM + col] + pos[row] * posw[col] + posb[col];
                C[(size_t)row * NDIM + col] = v;
            }
        }
    }
}

// ---------- dual-output WMMA GEMM: two GEMMs sharing the same A operand ----------
// C0 = A @ W0 (+bias0), C1 = A @ W1 (+bias1). One A fragment feeds two WMMAs.
template <int KDIM, int NDIM>
__global__ __launch_bounds__(128) void wmma_gemm_dual_kernel(
    const float* __restrict__ A,
    const __bf16* __restrict__ Wp0, const float* __restrict__ bias0, float* __restrict__ C0,
    const __bf16* __restrict__ Wp1, const float* __restrict__ bias1, float* __restrict__ C1,
    int M)
{
    constexpr int NKB = KDIM / 32;
    __shared__ __align__(32) __bf16 As[NKB * 32 * 16];
    const int tid = threadIdx.x;
    const int rowBase = blockIdx.x * 16;

    stage_A<KDIM>(A, As, rowBase, M, tid);
    __syncthreads();

    const int wid  = tid >> 5;
    const int lane = tid & 31;
    const int lr   = lane & 15;
    const int lh   = lane >> 4;
    const int ct   = blockIdx.y * 4 + wid;
    const int col  = ct * 16 + lr;

    const __bf16* wp0 = Wp0 + ((size_t)ct * NKB * 32 + lane) * 16;
    const __bf16* wp1 = Wp1 + ((size_t)ct * NKB * 32 + lane) * 16;

    v8f acc0 = {}, acc1 = {};
#pragma unroll
    for (int kb = 0; kb < NKB; ++kb) {
        v16bf a  = *(const v16bf*)&As[(kb * 32 + lane) * 16];
        v16bf b0 = *(const v16bf*)&wp0[(size_t)kb * 32 * 16];
        v16bf b1 = *(const v16bf*)&wp1[(size_t)kb * 32 * 16];
        acc0 = __builtin_amdgcn_wmma_f32_16x16x32_bf16(
            false, a, false, b0, (short)0, acc0, false, false);
        acc1 = __builtin_amdgcn_wmma_f32_16x16x32_bf16(
            false, a, false, b1, (short)0, acc1, false, false);
    }

    const float b0c = bias0 ? bias0[col] : 0.0f;
    const float b1c = bias1 ? bias1[col] : 0.0f;
    if (rowBase + 16 <= M) {
#pragma unroll
        for (int r = 0; r < 8; ++r) {
            int row = rowBase + r + 8 * lh;
            C0[(size_t)row * NDIM + col] = acc0[r] + b0c;
            C1[(size_t)row * NDIM + col] = acc1[r] + b1c;
        }
    } else {
#pragma unroll
        for (int r = 0; r < 8; ++r) {
            int row = rowBase + r + 8 * lh;
            if (row < M) {
                C0[(size_t)row * NDIM + col] = acc0[r] + b0c;
                C1[(size_t)row * NDIM + col] = acc1[r] + b1c;
            }
        }
    }
}

// ---------- tiny per-type attention tables ----------
__global__ void gat_tables_kernel(const float* __restrict__ lin_edge_w,
                                  const float* __restrict__ att_edge,
                                  const float* __restrict__ edge_emb,
                                  float* __restrict__ T, float* __restrict__ Atab,
                                  int H, int HCdim)
{
    int t = threadIdx.x;
    if (t >= 4 * H) return;
    int j = t / H, h = t % H;
    float tj[4];
#pragma unroll
    for (int jj = 0; jj < 4; ++jj) {
        float s = 0.f;
        for (int c = 0; c < 64; ++c)
            s += lin_edge_w[jj * HCdim + h * 64 + c] * att_edge[h * 64 + c];
        tj[jj] = s;
    }
    T[j * H + h] = tj[j];
    float s = 0.f;
#pragma unroll
    for (int jj = 0; jj < 4; ++jj) s += edge_emb[j * 4 + jj] * tj[jj];
    Atab[j * H + h] = s;
}

// ---------- edge pass 0: in-degree + segment-sum of 4d type embeddings ----------
__global__ void edge_count_kernel(const int* __restrict__ dst, const int* __restrict__ et,
                                  const float* __restrict__ emb1, const float* __restrict__ emb2,
                                  float* __restrict__ cnt, float* __restrict__ ls1,
                                  float* __restrict__ ls2, int E)
{
    int e = blockIdx.x * blockDim.x + threadIdx.x;
    if (e >= E) return;
    int d = dst[e], t = et[e];
    atomicAdd(&cnt[d], 1.0f);
#pragma unroll
    for (int j = 0; j < 4; ++j) {
        atomicAdd(&ls1[d * 4 + j], emb1[t * 4 + j]);
        atomicAdd(&ls2[d * 4 + j], emb2[t * 4 + j]);
    }
}

// ---------- per-(node,head) attention scalars + self-loop alpha ----------
template <int H>
__global__ void node_prep_kernel(const float* __restrict__ xs,
                                 const float* __restrict__ att_src,
                                 const float* __restrict__ att_dst,
                                 const float* __restrict__ T,
                                 const float* __restrict__ cnt,
                                 const float* __restrict__ loopsum,
                                 float* __restrict__ a_src, float* __restrict__ a_dst,
                                 float* __restrict__ lalpha, unsigned* __restrict__ amax,
                                 float* __restrict__ denom, int n)
{
    int i = blockIdx.x * blockDim.x + threadIdx.x;
    if (i >= n * H) return;
    int node = i / H, h = i % H;
    const float4* xrow = (const float4*)(xs + (size_t)node * (H * 64) + h * 64);
    const float4* as4 = (const float4*)(att_src + h * 64);
    const float4* ad4 = (const float4*)(att_dst + h * 64);
    float s = 0.f, dd = 0.f;
#pragma unroll 4
    for (int c = 0; c < 16; ++c) {
        float4 xv = xrow[c], a = as4[c], d = ad4[c];
        s  += xv.x * a.x + xv.y * a.y + xv.z * a.z + xv.w * a.w;
        dd += xv.x * d.x + xv.y * d.y + xv.z * d.z + xv.w * d.w;
    }
    float inv = 1.0f / fmaxf(cnt[node], 1.0f);
    float ae = 0.f;
#pragma unroll
    for (int j = 0; j < 4; ++j) ae += loopsum[node * 4 + j] * inv * T[j * H + h];
    float al = lrelu(s + dd + ae);
    a_src[i] = s; a_dst[i] = dd; lalpha[i] = al;
    amax[i] = f2ord(al);
    denom[i] = 0.f;
}

// ---------- edge pass 1: segment max of alpha (ordered-uint atomicMax) ----------
template <int H>
__global__ void edge_max_kernel(const int* __restrict__ src, const int* __restrict__ dst,
                                const int* __restrict__ et,
                                const float* __restrict__ a_src, const float* __restrict__ a_dst,
                                const float* __restrict__ Atab, unsigned* __restrict__ amax, int E)
{
    int i = blockIdx.x * blockDim.x + threadIdx.x;
    if (i >= E * H) return;
    int e = i / H, h = i - e * H;
    int s = src[e], d = dst[e], t = et[e];
    float al = lrelu(a_src[s * H + h] + a_dst[d * H + h] + Atab[t * H + h]);
    atomicMax(&amax[d * H + h], f2ord(al));
}

// ---------- edge pass 2: denom + unnormalized weighted message scatter ----------
template <int H>
__global__ void edge_agg_kernel(const int* __restrict__ src, const int* __restrict__ dst,
                                const int* __restrict__ et,
                                const float* __restrict__ a_src, const float* __restrict__ a_dst,
                                const float* __restrict__ Atab, const unsigned* __restrict__ amax,
                                const float* __restrict__ xs, float* __restrict__ denom,
                                float* __restrict__ out_un, int E)
{
    int i = blockIdx.x * blockDim.x + threadIdx.x;
    if (i >= E * H) return;
    int e = i / H, h = i - e * H;
    int s = src[e], d = dst[e], t = et[e];
    float al = lrelu(a_src[s * H + h] + a_dst[d * H + h] + Atab[t * H + h]);
    float ea = __expf(al - ord2f(amax[d * H + h]));
    atomicAdd(&denom[d * H + h], ea);
    const float4* xr4 = (const float4*)(xs + (size_t)s * (H * 64) + h * 64);
    float* orow = out_un + (size_t)d * (H * 64) + h * 64;
#pragma unroll 4
    for (int c4 = 0; c4 < 16; ++c4) {
        float4 xv = xr4[c4];
        atomicAdd(&orow[c4 * 4 + 0], xv.x * ea);
        atomicAdd(&orow[c4 * 4 + 1], xv.y * ea);
        atomicAdd(&orow[c4 * 4 + 2], xv.z * ea);
        atomicAdd(&orow[c4 * 4 + 3], xv.w * ea);
    }
}

// ---------- finalize: self-loop term, normalize, bias+residual, LayerNorm ----------
template <int HCD, int H, bool RELU>
__global__ __launch_bounds__(HCD) void node_finalize_kernel(
    const float* __restrict__ xs, const float* __restrict__ out_un,
    const float* __restrict__ resid, const float* __restrict__ bias,
    const float* __restrict__ lalpha, const unsigned* __restrict__ amax,
    const float* __restrict__ denom, const float* __restrict__ ln_g,
    const float* __restrict__ ln_b, float* __restrict__ outbuf, int n)
{
    __shared__ float s_den[H], s_eal[H];
    __shared__ float s_red[HCD / 32];
    __shared__ float s_stat[2];
    const int node = blockIdx.x;
    const int tid  = threadIdx.x;
    const int h    = tid >> 6;
    if (tid < H) {
        float am  = ord2f(amax[node * H + tid]);
        float eal = __expf(lalpha[node * H + tid] - am);
        s_eal[tid] = eal;
        s_den[tid] = denom[node * H + tid] + eal + 1e-16f;
    }
    __syncthreads();
    const size_t off = (size_t)node * HCD + tid;
    float v = (out_un[off] + xs[off] * s_eal[h]) / s_den[h];
    v += bias[tid] + resid[off];

    float sum = v;
#pragma unroll
    for (int o = 16; o > 0; o >>= 1) sum += __shfl_down(sum, o);
    if ((tid & 31) == 0) s_red[tid >> 5] = sum;
    __syncthreads();
    if (tid == 0) {
        float tot = 0.f;
        for (int w = 0; w < HCD / 32; ++w) tot += s_red[w];
        s_stat[0] = tot / (float)HCD;
    }
    __syncthreads();
    const float mu = s_stat[0];
    float dv = v - mu;
    float sq = dv * dv;
#pragma unroll
    for (int o = 16; o > 0; o >>= 1) sq += __shfl_down(sq, o);
    if ((tid & 31) == 0) s_red[tid >> 5] = sq;
    __syncthreads();
    if (tid == 0) {
        float tot = 0.f;
        for (int w = 0; w < HCD / 32; ++w) tot += s_red[w];
        s_stat[1] = tot / (float)HCD;
    }
    __syncthreads();
    float y = dv * rsqrtf(s_stat[1] + 1e-5f) * ln_g[tid] + ln_b[tid];
    if (RELU) y = y > 0.f ? y : 0.f;
    outbuf[off] = y;
}

// ---------- classifier: gather + 64x3 matmul ----------
__global__ void classifier_kernel(const float* __restrict__ hid, const int* __restrict__ idx,
                                  const float* __restrict__ w, const float* __restrict__ b,
                                  float* __restrict__ out, int n)
{
    int i = blockIdx.x * blockDim.x + threadIdx.x;
    if (i >= n) return;
    const float* x = hid + (size_t)idx[i] * 64;
    float o0 = b[0], o1 = b[1], o2 = b[2];
    for (int c = 0; c < 64; ++c) {
        float xv = x[c];
        o0 += xv * w[c * 3 + 0];
        o1 += xv * w[c * 3 + 1];
        o2 += xv * w[c * 3 + 2];
    }
    out[i * 3 + 0] = o0; out[i * 3 + 1] = o1; out[i * 3 + 2] = o2;
}

// =====================================================================
extern "C" void kernel_launch(void* const* d_in, const int* in_sizes, int n_in,
                              void* d_out, int out_size, void* d_ws, size_t ws_size,
                              hipStream_t stream)
{
    (void)in_sizes; (void)n_in; (void)out_size; (void)ws_size;
    const float* features  = (const float*)d_in[0];
    const int*   edge_idx  = (const int*)  d_in[1];
    const int*   aspect    = (const int*)  d_in[2];
    const int*   etypes    = (const int*)  d_in[3];
    const int*   ntypes    = (const int*)  d_in[4];
    const float* positions = (const float*)d_in[5];
    const float* fe_w1 = (const float*)d_in[6];   const float* fe_b1 = (const float*)d_in[7];
    const float* fe_w2 = (const float*)d_in[8];   const float* fe_b2 = (const float*)d_in[9];
    const float* fe_pos_w = (const float*)d_in[10]; const float* fe_pos_b = (const float*)d_in[11];
    const float* fe_type_emb = (const float*)d_in[12];
    const float* fe_out_w = (const float*)d_in[13]; const float* fe_out_b = (const float*)d_in[14];
    const float* g1_lin_w = (const float*)d_in[15];
    const float* g1_att_src = (const float*)d_in[16]; const float* g1_att_dst = (const float*)d_in[17];
    const float* g1_edge_emb = (const float*)d_in[18]; const float* g1_lin_edge_w = (const float*)d_in[19];
    const float* g1_att_edge = (const float*)d_in[20];
    const float* g1_bias = (const float*)d_in[21];
    const float* g1_res_w = (const float*)d_in[22]; const float* g1_res_b = (const float*)d_in[23];
    const float* g1_ln_g = (const float*)d_in[24];  const float* g1_ln_b = (const float*)d_in[25];
    const float* g2_lin_w = (const float*)d_in[26];
    const float* g2_att_src = (const float*)d_in[27]; const float* g2_att_dst = (const float*)d_in[28];
    const float* g2_edge_emb = (const float*)d_in[29]; const float* g2_lin_edge_w = (const float*)d_in[30];
    const float* g2_att_edge = (const float*)d_in[31];
    const float* g2_bias = (const float*)d_in[32];
    const float* g2_res_w = (const float*)d_in[33]; const float* g2_res_b = (const float*)d_in[34];
    const float* g2_ln_g = (const float*)d_in[35];  const float* g2_ln_b = (const float*)d_in[36];
    const float* cls_w = (const float*)d_in[37];    const float* cls_b = (const float*)d_in[38];

    const int* src = edge_idx;
    const int* dst = edge_idx + NE;

    // ---- carve workspace ----
    char* p = (char*)d_ws;
    auto alloc = [&](size_t bytes) -> char* {
        char* r = p; p += (bytes + 255) & ~(size_t)255; return r;
    };
    float* h1     = (float*)alloc((size_t)NN * HDIM * 4);  // reused as xs2
    float* h2     = (float*)alloc((size_t)NN * HDIM * 4);  // reused as resid2
    float* hfe    = (float*)alloc((size_t)NN * HDIM * 4);  // reused as out_un2/hid2
    float* xs1    = (float*)alloc((size_t)NN * HC1 * 4);
    float* resid1 = (float*)alloc((size_t)NN * HC1 * 4);
    float* outun1 = (float*)alloc((size_t)NN * HC1 * 4);   // hid1 aliases this
    float* asrc1  = (float*)alloc((size_t)NN * NH1 * 4);
    float* adst1  = (float*)alloc((size_t)NN * NH1 * 4);
    float* lal1   = (float*)alloc((size_t)NN * NH1 * 4);
    unsigned* am1 = (unsigned*)alloc((size_t)NN * NH1 * 4);
    float* den1   = (float*)alloc((size_t)NN * NH1 * 4);
    float* asrc2  = (float*)alloc((size_t)NN * 4);
    float* adst2  = (float*)alloc((size_t)NN * 4);
    float* lal2   = (float*)alloc((size_t)NN * 4);
    unsigned* am2 = (unsigned*)alloc((size_t)NN * 4);
    float* den2   = (float*)alloc((size_t)NN * 4);
    float* cnt    = (float*)alloc((size_t)NN * 4);
    float* ls1    = (float*)alloc((size_t)NN * 4 * 4);
    float* ls2    = (float*)alloc((size_t)NN * 4 * 4);
    float* T1     = (float*)alloc(16 * 4);
    float* Atab1  = (float*)alloc(16 * 4);
    float* T2     = (float*)alloc(4 * 4);
    float* Atab2  = (float*)alloc(4 * 4);
    // bf16 fragment-packed weights
    __bf16* fe_w1p  = (__bf16*)alloc((size_t)DIN * HDIM * 2);
    __bf16* fe_w2p  = (__bf16*)alloc((size_t)HDIM * HDIM * 2);
    __bf16* fe_outp = (__bf16*)alloc((size_t)HDIM * HDIM * 2);
    __bf16* g1_linp = (__bf16*)alloc((size_t)HDIM * HC1 * 2);
    __bf16* g1_resp = (__bf16*)alloc((size_t)HDIM * HC1 * 2);
    __bf16* g2_linp = (__bf16*)alloc((size_t)HC1 * HDIM * 2);
    __bf16* g2_resp = (__bf16*)alloc((size_t)HC1 * HDIM * 2);

    float* xs2    = h1;
    float* resid2 = h2;
    float* outun2 = hfe;   // reused after hfe is consumed
    float* hid1   = outun1;
    float* hid2   = outun2;

    const dim3 blk128(128);
    const dim3 gRow((NN + 15) / 16, 1);
    const dim3 gRow4((NN + 15) / 16, 4);

    // ---- zero accumulators ----
    hipMemsetAsync(cnt, 0, (size_t)NN * 4, stream);
    hipMemsetAsync(ls1, 0, (size_t)NN * 16, stream);
    hipMemsetAsync(ls2, 0, (size_t)NN * 16, stream);
    hipMemsetAsync(outun1, 0, (size_t)NN * HC1 * 4, stream);

    // ---- pack weights (tiny) ----
    auto pack = [&](const float* W, __bf16* Wp, int kd, int nd) {
        int tot = kd * nd;
        pack_weights_kernel<<<(tot + 255) / 256, 256, 0, stream>>>(W, Wp, kd, nd);
    };
    pack(fe_w1,   fe_w1p,  DIN,  HDIM);
    pack(fe_w2,   fe_w2p,  HDIM, HDIM);
    pack(fe_out_w,fe_outp, HDIM, HDIM);
    pack(g1_lin_w,g1_linp, HDIM, HC1);
    pack(g1_res_w,g1_resp, HDIM, HC1);
    pack(g2_lin_w,g2_linp, HC1,  HDIM);
    pack(g2_res_w,g2_resp, HC1,  HDIM);

    // ---- attention type tables ----
    gat_tables_kernel<<<1, 32, 0, stream>>>(g1_lin_edge_w, g1_att_edge, g1_edge_emb, T1, Atab1, NH1, HC1);
    gat_tables_kernel<<<1, 32, 0, stream>>>(g2_lin_edge_w, g2_att_edge, g2_edge_emb, T2, Atab2, 1, HDIM);

    // ---- feature extractor (3 WMMA GEMMs) ----
    wmma_gemm_kernel<DIN, HDIM, 1><<<gRow, blk128, 0, stream>>>(
        features, fe_w1p, fe_b1, h1, NN, nullptr, nullptr, nullptr, nullptr, nullptr);
    wmma_gemm_kernel<HDIM, HDIM, 2><<<gRow, blk128, 0, stream>>>(
        h1, fe_w2p, fe_b2, h2, NN, ntypes, fe_type_emb, positions, fe_pos_w, fe_pos_b);
    wmma_gemm_kernel<HDIM, HDIM, 0><<<gRow, blk128, 0, stream>>>(
        h2, fe_outp, fe_out_b, hfe, NN, nullptr, nullptr, nullptr, nullptr, nullptr);

    // ---- degree / loop-attr sums ----
    edge_count_kernel<<<(NE + 255) / 256, 256, 0, stream>>>(
        dst, etypes, g1_edge_emb, g2_edge_emb, cnt, ls1, ls2, NE);

    // ---- GAT layer 1: fused lin + residual GEMM (shared A = hfe) ----
    wmma_gemm_dual_kernel<HDIM, HC1><<<gRow4, blk128, 0, stream>>>(
        hfe, g1_linp, nullptr, xs1, g1_resp, g1_res_b, resid1, NN);
    node_prep_kernel<NH1><<<(NN * NH1 + 255) / 256, 256, 0, stream>>>(
        xs1, g1_att_src, g1_att_dst, T1, cnt, ls1, asrc1, adst1, lal1, am1, den1, NN);
    edge_max_kernel<NH1><<<(NE * NH1 + 255) / 256, 256, 0, stream>>>(
        src, dst, etypes, asrc1, adst1, Atab1, am1, NE);
    edge_agg_kernel<NH1><<<(NE * NH1 + 255) / 256, 256, 0, stream>>>(
        src, dst, etypes, asrc1, adst1, Atab1, am1, xs1, den1, outun1, NE);
    node_finalize_kernel<HC1, NH1, true><<<NN, HC1, 0, stream>>>(
        xs1, outun1, resid1, g1_bias, lal1, am1, den1, g1_ln_g, g1_ln_b, hid1, NN);

    // ---- GAT layer 2 (H=1): fused lin + residual GEMM (shared A = hid1) ----
    wmma_gemm_dual_kernel<HC1, HDIM><<<gRow, blk128, 0, stream>>>(
        hid1, g2_linp, nullptr, xs2, g2_resp, g2_res_b, resid2, NN);
    hipMemsetAsync(outun2, 0, (size_t)NN * HDIM * 4, stream);  // hfe no longer needed
    node_prep_kernel<1><<<(NN + 255) / 256, 256, 0, stream>>>(
        xs2, g2_att_src, g2_att_dst, T2, cnt, ls2, asrc2, adst2, lal2, am2, den2, NN);
    edge_max_kernel<1><<<(NE + 255) / 256, 256, 0, stream>>>(
        src, dst, etypes, asrc2, adst2, Atab2, am2, NE);
    edge_agg_kernel<1><<<(NE + 255) / 256, 256, 0, stream>>>(
        src, dst, etypes, asrc2, adst2, Atab2, am2, xs2, den2, outun2, NE);
    node_finalize_kernel<HDIM, 1, false><<<NN, HDIM, 0, stream>>>(
        xs2, outun2, resid2, g2_bias, lal2, am2, den2, g2_ln_g, g2_ln_b, hid2, NN);

    // ---- classifier ----
    classifier_kernel<<<(4096 + 255) / 256, 256, 0, stream>>>(
        hid2, aspect, cls_w, cls_b, (float*)d_out, 4096);
}
